// Mamba2_39462159515683
// MI455X (gfx1250) — matrix operand order repaired and checked by reading, not compiled, not run on hardware
//
#include <hip/hip_runtime.h>

// ---------------- problem constants ----------------
#define D_MODEL   1024
#define D_INNER   2048
#define N_HEADS   32
#define D_HEAD    64
#define D_STATE   64
#define D_CONV    4
#define CHUNK     256
#define CONV_DIM  (D_INNER + 2 * D_STATE)               // 2176
#define D_IN_PROJ (2 * D_INNER + 2 * D_STATE + N_HEADS) // 4256
#define B_SZ      2
#define L_SEQ     4096
#define T_TOK     (B_SZ * L_SEQ)                        // 8192
#define NC        (L_SEQ / CHUNK)                       // 16
#define EPSF      1e-5f

typedef __attribute__((ext_vector_type(16))) __bf16 v16bf;
typedef __attribute__((ext_vector_type(8)))  float  v8f;

#if defined(__has_builtin)
#if __has_builtin(__builtin_amdgcn_tensor_load_to_lds) && __has_builtin(__builtin_amdgcn_s_wait_tensorcnt)
#define HAVE_TDM 1
#endif
#endif
#ifndef HAVE_TDM
#define HAVE_TDM 0
#endif

// ---------------- bf16 helpers (bit-exact RNE) ----------------
__device__ __forceinline__ __bf16 f2bf(float f) {
  unsigned u = __float_as_uint(f);
  u += 0x7fffu + ((u >> 16) & 1u);
  unsigned short h = (unsigned short)(u >> 16);
  __bf16 r;
  __builtin_memcpy(&r, &h, sizeof(r));
  return r;
}
__device__ __forceinline__ float bf2f(__bf16 b) {
  unsigned short h;
  __builtin_memcpy(&h, &b, sizeof(h));
  return __uint_as_float(((unsigned)h) << 16);
}
__device__ __forceinline__ v8f v8f_zero() {
  v8f z;
#pragma unroll
  for (int i = 0; i < 8; ++i) z[i] = 0.f;
  return z;
}

// ---------------- WMMA fragment loaders (contiguous k = 16*(lane/16) + e, same order for A and B) ----------------
__device__ __forceinline__ v16bf frag_a_16x32(const __bf16* src, int ld, int lane) {
  const int m = lane & 15, kh = lane >> 4;
  const __bf16* r = src + m * ld + kh * 16;
  v16bf f;
#pragma unroll
  for (int e = 0; e < 16; ++e) f[e] = r[e];
  return f;
}
__device__ __forceinline__ v16bf frag_b_kxn(const __bf16* src, int ld, int lane) {
  const int n = lane & 15, kh = lane >> 4;
  v16bf f;
#pragma unroll
  for (int e = 0; e < 16; ++e) f[e] = src[(kh * 16 + e) * ld + n];
  return f;
}
__device__ __forceinline__ v16bf frag_b_nxk(const __bf16* src, int ld, int lane) {
  const int n = lane & 15, kh = lane >> 4;
  const __bf16* r = src + n * ld + kh * 16;
  v16bf f;
#pragma unroll
  for (int e = 0; e < 16; ++e) f[e] = r[e];
  return f;
}
__device__ __forceinline__ v8f wmma_bf16(v16bf a, v16bf b, v8f c) {
  return __builtin_amdgcn_wmma_f32_16x16x32_bf16(false, a, false, b, (short)0, c, false, false);
}

// ---------------- Tensor Data Mover: 2D bf16 tile (64 rows x 32 cols) -> LDS ----------------
#if HAVE_TDM
typedef __attribute__((ext_vector_type(4))) unsigned int u32x4;
typedef __attribute__((ext_vector_type(8))) int i32x8;
typedef __attribute__((ext_vector_type(4))) int i32x4;

// D# per CDNA5 ISA ch.8: group0 {count=1, lds_addr, global_addr, type=2},
// group1 {data_size=2B, tensor dims (clamped remaining extent -> OOB rows read as 0),
//         tile 32x64, dim0 stride}. Groups 2/3 zero (2D tensor).
__device__ __forceinline__ void tdm_load_bf16_64x32(unsigned lds_off, const __bf16* gptr,
                                                    int row_stride_elts, int tensor_rows,
                                                    int tensor_k) {
  const unsigned long long ga = (unsigned long long)(const void*)gptr;
  u32x4 g0;
  g0[0] = 1u;                                              // count=1, user D#
  g0[1] = lds_off;                                         // lds_addr (dyn-LDS base 0)
  g0[2] = (unsigned)(ga & 0xffffffffu);                    // global_addr[31:0]
  g0[3] = (unsigned)((ga >> 32) & 0x01ffffffu) | (2u << 30); // addr[56:32] | type=2
  i32x8 g1;
  g1[0] = (1 << 16);                                       // data_size=1 (2 bytes)
  g1[1] = (tensor_k & 0xffff) << 16;                       // tensor_dim0[15:0]
  g1[2] = ((unsigned)tensor_k >> 16) | ((tensor_rows & 0xffff) << 16); // dim0 hi | dim1 lo
  g1[3] = ((unsigned)tensor_rows >> 16) | (32 << 16);      // dim1 hi | tile_dim0=32
  g1[4] = 64;                                              // tile_dim1=64, tile_dim2=0
  g1[5] = row_stride_elts;                                 // tensor_dim0_stride[31:0]
  g1[6] = 0;                                               // stride hi, dim1_stride lo
  g1[7] = 0;
  i32x4 z4;
  z4[0] = z4[1] = z4[2] = z4[3] = 0;
  i32x8 z8;
#pragma unroll
  for (int i = 0; i < 8; ++i) z8[i] = 0;
  __builtin_amdgcn_tensor_load_to_lds(g0, g1, z4, z4, z8, 0);
}
#endif

// ---------------- elementwise kernels ----------------
__global__ void k_cast_bf16(const float* __restrict__ src, __bf16* __restrict__ dst, int n) {
  const int i = blockIdx.x * 256 + threadIdx.x;
  if (i < n) dst[i] = f2bf(src[i]);
}

__global__ void k_rmsnorm_bf16(const float* __restrict__ x, const float* __restrict__ w,
                               __bf16* __restrict__ out, int D) {
  const int t = blockIdx.x;
  const float* row = x + (size_t)t * D;
  __shared__ float red[256];
  float ss = 0.f;
  for (int i = threadIdx.x; i < D; i += 256) { const float v = row[i]; ss += v * v; }
  red[threadIdx.x] = ss;
  __syncthreads();
  for (int s = 128; s > 0; s >>= 1) {
    if (threadIdx.x < s) red[threadIdx.x] += red[threadIdx.x + s];
    __syncthreads();
  }
  const float scale = rsqrtf(red[0] / (float)D + EPSF);
  for (int i = threadIdx.x; i < D; i += 256)
    out[(size_t)t * D + i] = f2bf(row[i] * scale * w[i]);
}

__global__ void k_gate_rmsnorm_bf16(const float* __restrict__ y, const float* __restrict__ zx,
                                    const float* __restrict__ gw, __bf16* __restrict__ out) {
  const int t = blockIdx.x;
  const float* yr = y + (size_t)t * D_INNER;
  const float* zr = zx + (size_t)t * D_IN_PROJ; // z occupies cols [0, D_INNER)
  __shared__ float red[256];
  float ss = 0.f;
  for (int i = threadIdx.x; i < D_INNER; i += 256) {
    const float zv = zr[i];
    const float v = yr[i] * (zv / (1.f + expf(-zv)));
    ss += v * v;
  }
  red[threadIdx.x] = ss;
  __syncthreads();
  for (int s = 128; s > 0; s >>= 1) {
    if (threadIdx.x < s) red[threadIdx.x] += red[threadIdx.x + s];
    __syncthreads();
  }
  const float scale = rsqrtf(red[0] / (float)D_INNER + EPSF);
  for (int i = threadIdx.x; i < D_INNER; i += 256) {
    const float zv = zr[i];
    const float v = yr[i] * (zv / (1.f + expf(-zv)));
    out[(size_t)t * D_INNER + i] = f2bf(v * scale * gw[i]);
  }
}

__global__ void k_conv_silu(const float* __restrict__ zx, const float* __restrict__ cw,
                            const float* __restrict__ cb, float* __restrict__ xbc) {
  const int c = blockIdx.x * 256 + threadIdx.x;
  if (c >= CONV_DIM) return;
  const int t = blockIdx.y;
  const int l = t & (L_SEQ - 1);
  float acc = cb[c];
#pragma unroll
  for (int i = 0; i < D_CONV; ++i) {
    const int ls = l - (D_CONV - 1) + i;
    if (ls >= 0)
      acc += zx[(size_t)(t - (D_CONV - 1) + i) * D_IN_PROJ + D_INNER + c] * cw[c * D_CONV + i];
  }
  acc = acc / (1.f + expf(-acc)); // silu
  xbc[(size_t)t * CONV_DIM + c] = acc;
}

__global__ void k_dt(const float* __restrict__ zx, const float* __restrict__ dtb,
                     float* __restrict__ dt) {
  const int idx = blockIdx.x * 256 + threadIdx.x;
  if (idx >= T_TOK * N_HEADS) return;
  const int t = idx >> 5, h = idx & 31;
  const float v = zx[(size_t)t * D_IN_PROJ + (D_INNER + CONV_DIM) + h] + dtb[h];
  dt[idx] = (v > 20.f) ? v : log1pf(expf(v));
}

__global__ void k_cumsum(const float* __restrict__ dt, const float* __restrict__ A_log,
                         float* __restrict__ dAcs, float* __restrict__ dAlast) {
  const int z = blockIdx.x, h = blockIdx.y, b = blockIdx.z;
  const int l = threadIdx.x;
  const int t = b * L_SEQ + z * CHUNK + l;
  const float Av = -expf(A_log[h]);
  __shared__ float sh[CHUNK];
  sh[l] = dt[(size_t)t * N_HEADS + h] * Av;
  __syncthreads();
  for (int off = 1; off < CHUNK; off <<= 1) {
    const float add = (l >= off) ? sh[l - off] : 0.f;
    __syncthreads();
    sh[l] += add;
    __syncthreads();
  }
  const size_t base = (((size_t)b * N_HEADS + h) * NC + z) * CHUNK;
  dAcs[base + l] = sh[l];
  if (l == CHUNK - 1) dAlast[((size_t)b * N_HEADS + h) * NC + z] = sh[l];
}

// ---------------- bf16 WMMA GEMM: C[m][n] = sum_k A[m][k]*W[n][k] (+resid) ----------------
// BM=BN=64, BK=32; 256 threads = 8 waves. Double-buffered LDS stages (2 x 8KB dynamic),
// fed by the Tensor Data Mover when available (wave 0 issues, TENSORcnt sync), else by
// one 128-bit load per thread per tile.
__device__ __forceinline__ void gemm_issue_stage(const __bf16* __restrict__ A,
                                                 const __bf16* __restrict__ W,
                                                 char* smem, int stage, int m0, int n0, int k0,
                                                 int M, int N, int K, int tid, int wid) {
#if HAVE_TDM
  if (wid == 0) {
    tdm_load_bf16_64x32((unsigned)(stage * 8192),
                        A + (size_t)m0 * K + k0, K, M - m0, K - k0);
    tdm_load_bf16_64x32((unsigned)(stage * 8192 + 4096),
                        W + (size_t)n0 * K + k0, K, N - n0, K - k0);
  }
  (void)tid; (void)smem;
#else
  __bf16* As = (__bf16*)(smem + stage * 8192);
  __bf16* Ws = As + 64 * 32;
  const int r = tid >> 2, c = (tid & 3) * 8; // 8 bf16 = 16B per thread
  *(uint4*)(As + r * 32 + c) = *(const uint4*)(A + (size_t)(m0 + r) * K + k0 + c);
  uint4 wv = {0u, 0u, 0u, 0u};
  if (n0 + r < N) wv = *(const uint4*)(W + (size_t)(n0 + r) * K + k0 + c);
  *(uint4*)(Ws + r * 32 + c) = wv;
  (void)wid; (void)M;
#endif
}

__global__ void k_gemm_bf16(const __bf16* __restrict__ A, const __bf16* __restrict__ W,
                            const float* __restrict__ resid, float* __restrict__ C,
                            int M, int N, int K) {
  extern __shared__ char smem[]; // 2 stages x (A 64x32 + W 64x32) bf16 = 16KB, base off 0
  const int n0 = blockIdx.x * 64;
  const int m0 = blockIdx.y * 64;
  const int tid = threadIdx.x;
  const int wid = tid >> 5, lane = tid & 31;
  const int wm = wid & 3, wn = wid >> 2;
  v8f acc0 = v8f_zero(), acc1 = v8f_zero();

  const int nst = K / 32;
  gemm_issue_stage(A, W, smem, 0, m0, n0, 0, M, N, K, tid, wid);
  for (int ks = 0; ks < nst; ++ks) {
    if (ks + 1 < nst)
      gemm_issue_stage(A, W, smem, (ks + 1) & 1, m0, n0, (ks + 1) * 32, M, N, K, tid, wid);
#if HAVE_TDM
    if (wid == 0) {
      if (ks + 1 < nst) {
        __builtin_amdgcn_s_wait_tensorcnt(2); // stage ks done; next stage stays in flight
      } else {
        __builtin_amdgcn_s_wait_tensorcnt(0); // drain
      }
    }
#endif
    __syncthreads();
    const __bf16* As = (const __bf16*)(smem + (ks & 1) * 8192);
    const __bf16* Ws = As + 64 * 32;
    const v16bf a = frag_a_16x32(As + (wm * 16) * 32, 32, lane);
    const v16bf b0 = frag_b_nxk(Ws + ((wn * 2 + 0) * 16) * 32, 32, lane);
    acc0 = wmma_bf16(a, b0, acc0);
    const v16bf b1 = frag_b_nxk(Ws + ((wn * 2 + 1) * 16) * 32, 32, lane);
    acc1 = wmma_bf16(a, b1, acc1);
    __syncthreads();
  }

  const int nlane = lane & 15, mh = lane >> 4;
#pragma unroll
  for (int j = 0; j < 2; ++j) {
    const v8f acc = j ? acc1 : acc0;
    const int n = n0 + (wn * 2 + j) * 16 + nlane;
    if (n < N) {
#pragma unroll
      for (int r = 0; r < 8; ++r) {
        const int m = m0 + wm * 16 + mh * 8 + r;
        float v = acc[r];
        if (resid) v += resid[(size_t)m * N + n];
        C[(size_t)m * N + n] = v;
      }
    }
  }
}

// ---------------- chunk states: states[p][n] = sum_l Xdt[l][p]*decay[l]*Bc[l][n] ----------------
__global__ void k_chunk_states(const float* __restrict__ xbc, const float* __restrict__ dt,
                               const float* __restrict__ dAcs, const float* __restrict__ dAlast,
                               float* __restrict__ states) {
  extern __shared__ char smem[];
  __bf16* XdT = (__bf16*)smem;        // [p=64][l=256]
  __bf16* Bs  = XdT + D_HEAD * CHUNK; // [l=256][n=64]
  const int h = blockIdx.x, z = blockIdx.y, b = blockIdx.z;
  const int tid = threadIdx.x;
  const size_t cbase = ((size_t)b * N_HEADS + h) * NC + z;
  const float dAl = dAlast[cbase];
  const float* dacs = dAcs + cbase * CHUNK;

  for (int idx = tid; idx < CHUNK * D_HEAD; idx += 128) {
    const int l = idx >> 6, c = idx & 63;
    const int t = b * L_SEQ + z * CHUNK + l;
    const float decay = expf(dAl - dacs[l]);
    XdT[c * CHUNK + l] = f2bf(xbc[(size_t)t * CONV_DIM + h * D_HEAD + c] *
                              dt[(size_t)t * N_HEADS + h] * decay);
    Bs[l * D_STATE + c] = f2bf(xbc[(size_t)t * CONV_DIM + D_INNER + c]);
  }
  __syncthreads();

  const int wid = tid >> 5, lane = tid & 31;
  v8f acc[4];
#pragma unroll
  for (int i = 0; i < 4; ++i) acc[i] = v8f_zero();

  for (int ks = 0; ks < CHUNK; ks += 32) {
    const v16bf a = frag_a_16x32(XdT + (wid * 16) * CHUNK + ks, CHUNK, lane);
#pragma unroll
    for (int nt = 0; nt < 4; ++nt) {
      const v16bf bf = frag_b_kxn(Bs + ks * D_STATE + nt * 16, D_STATE, lane);
      acc[nt] = wmma_bf16(a, bf, acc[nt]);
    }
  }
  float* sout = states + (((size_t)b * NC + z) * N_HEADS + h) * (D_HEAD * D_STATE);
  const int n = lane & 15, mh = lane >> 4;
#pragma unroll
  for (int nt = 0; nt < 4; ++nt)
#pragma unroll
    for (int r = 0; r < 8; ++r)
      sout[(wid * 16 + mh * 8 + r) * D_STATE + nt * 16 + n] = acc[nt][r];
}

// ---------------- inter-chunk scan ----------------
__global__ void k_state_scan(const float* __restrict__ states, const float* __restrict__ dAlast,
                             float* __restrict__ prev) {
  const int h = blockIdx.x, b = blockIdx.y;
  const int off = threadIdx.x * 16;
  float run[16];
#pragma unroll
  for (int j = 0; j < 16; ++j) run[j] = 0.f;
  for (int z = 0; z < NC; ++z) {
    const size_t base = (((size_t)b * NC + z) * N_HEADS + h) * (D_HEAD * D_STATE) + off;
    const float cd = expf(dAlast[((size_t)b * N_HEADS + h) * NC + z]);
#pragma unroll
    for (int j = 0; j < 16; ++j) {
      prev[base + j] = run[j];
      run[j] = run[j] * cd + states[base + j];
    }
  }
}

// ---------------- chunk output: Y = Yd (masked-G WMMA) + Yo (prev WMMA) + D*X ----------------
__global__ void k_chunk_output(const float* __restrict__ xbc, const float* __restrict__ dt,
                               const float* __restrict__ dAcs, const float* __restrict__ prev,
                               const float* __restrict__ Dvec, float* __restrict__ Y) {
  extern __shared__ char smem[];
  __bf16* Cc = (__bf16*)smem;              // [l=256][n=64]
  __bf16* Bs = Cc + CHUNK * D_STATE;       // [s=256][n=64]
  __bf16* Xd = Bs + CHUNK * D_STATE;       // [s=256][p=64]
  __bf16* Pv = Xd + CHUNK * D_HEAD;        // [p=64][n=64]
  __bf16* Gs = Pv + D_HEAD * D_STATE;      // 8 waves x [16][32] scratch
  float*  dA = (float*)(Gs + 8 * 16 * 32); // [256]
  const int zb = blockIdx.x;               // b*NC + z
  const int b = zb / NC, z = zb % NC;
  const int h = blockIdx.y;
  const int tid = threadIdx.x;
  const size_t cbase = (((size_t)b * N_HEADS + h) * NC + z) * CHUNK;

  for (int idx = tid; idx < CHUNK * D_HEAD; idx += 256) {
    const int l = idx >> 6, c = idx & 63;
    const int t = b * L_SEQ + z * CHUNK + l;
    const float* row = xbc + (size_t)t * CONV_DIM;
    Xd[idx] = f2bf(row[h * D_HEAD + c] * dt[(size_t)t * N_HEADS + h]);
    Bs[l * D_STATE + c] = f2bf(row[D_INNER + c]);
    Cc[l * D_STATE + c] = f2bf(row[D_INNER + D_STATE + c]);
  }
  for (int idx = tid; idx < D_HEAD * D_STATE; idx += 256)
    Pv[idx] = f2bf(prev[((size_t)zb * N_HEADS + h) * (D_HEAD * D_STATE) + idx]);
  if (tid < CHUNK) dA[tid] = dAcs[cbase + tid];
  __syncthreads();

  const int wid = tid >> 5, lane = tid & 31;
  const int nlane = lane & 15, mh = lane >> 4;
  __bf16* gscr = Gs + wid * (16 * 32);
  const float Dv = Dvec[h];

  for (int pass = 0; pass < 2; ++pass) {
    const int lt = wid + pass * 8;
    const int l0 = lt * 16;
    v8f yacc[4];
#pragma unroll
    for (int i = 0; i < 4; ++i) yacc[i] = v8f_zero();

    // ---- Yo: (Cc[l,:]*exp(dAcs[l])) x prev[p][n]^T over K=n=64 ----
    const float eE = expf(dA[l0 + nlane]); // A-fragment row == lane&15
#pragma unroll
    for (int ks = 0; ks < 64; ks += 32) {
      v16bf a = frag_a_16x32(Cc + l0 * D_STATE + ks, D_STATE, lane);
#pragma unroll
      for (int e = 0; e < 16; ++e) a[e] = f2bf(bf2f(a[e]) * eE);
#pragma unroll
      for (int pt = 0; pt < 4; ++pt) {
        const v16bf bp = frag_b_nxk(Pv + (pt * 16) * D_STATE + ks, D_STATE, lane);
        yacc[pt] = wmma_bf16(a, bp, yacc[pt]);
      }
    }

    // ---- Yd: G = Cc x Bc^T, mask/decay, then G x Xdt ----
    const v16bf ac0 = frag_a_16x32(Cc + l0 * D_STATE + 0, D_STATE, lane);
    const v16bf ac1 = frag_a_16x32(Cc + l0 * D_STATE + 32, D_STATE, lane);
    const int npair = lt / 2 + 1;
    for (int sp = 0; sp < npair; ++sp) {
      const int s0 = sp * 32;
#pragma unroll
      for (int j = 0; j < 2; ++j) {
        v8f g = v8f_zero();
        const v16bf b0 = frag_b_nxk(Bs + (s0 + j * 16) * D_STATE + 0, D_STATE, lane);
        g = wmma_bf16(ac0, b0, g);
        const v16bf b1 = frag_b_nxk(Bs + (s0 + j * 16) * D_STATE + 32, D_STATE, lane);
        g = wmma_bf16(ac1, b1, g);
        const int s = s0 + j * 16 + nlane;
        const float dAs = dA[s];
#pragma unroll
        for (int r = 0; r < 8; ++r) {
          const int l = l0 + mh * 8 + r;
          const float gv = (s <= l) ? g[r] * expf(dA[l] - dAs) : 0.f;
          gscr[(mh * 8 + r) * 32 + j * 16 + nlane] = f2bf(gv);
        }
      }
      asm volatile("s_wait_dscnt 0x0" ::: "memory"); // intra-wave LDS RAW on gscr
      const v16bf ag = frag_a_16x32(gscr, 32, lane);
#pragma unroll
      for (int pt = 0; pt < 4; ++pt) {
        const v16bf bx = frag_b_kxn(Xd + s0 * D_HEAD + pt * 16, D_HEAD, lane);
        yacc[pt] = wmma_bf16(ag, bx, yacc[pt]);
      }
    }

    // ---- store: Y += D*X skip ----
#pragma unroll
    for (int pt = 0; pt < 4; ++pt) {
#pragma unroll
      for (int r = 0; r < 8; ++r) {
        const int l = l0 + mh * 8 + r;
        const int t = b * L_SEQ + z * CHUNK + l;
        const int p = pt * 16 + nlane;
        Y[(size_t)t * D_INNER + h * D_HEAD + p] =
            yacc[pt][r] + Dv * xbc[(size_t)t * CONV_DIM + h * D_HEAD + p];
      }
    }
  }
}

// ---------------- driver ----------------
extern "C" void kernel_launch(void* const* d_in, const int* in_sizes, int n_in,
                              void* d_out, int out_size, void* d_ws, size_t ws_size,
                              hipStream_t stream) {
  (void)in_sizes; (void)n_in; (void)out_size; (void)ws_size;
  const float* x       = (const float*)d_in[0];
  const float* in_w    = (const float*)d_in[1];
  const float* conv_w  = (const float*)d_in[2];
  const float* conv_b  = (const float*)d_in[3];
  const float* dt_bias = (const float*)d_in[4];
  const float* A_log   = (const float*)d_in[5];
  const float* Dp      = (const float*)d_in[6];
  const float* gnorm_w = (const float*)d_in[7];
  const float* out_w   = (const float*)d_in[8];
  const float* norm_w  = (const float*)d_in[9];

  char* p = (char*)d_ws;
  auto alloc = [&](size_t bytes) -> void* {
    void* r = (void*)p;
    p += (bytes + 255) & ~(size_t)255;
    return r;
  };
  __bf16* hb    = (__bf16*)alloc((size_t)T_TOK * D_MODEL * 2);
  __bf16* iwb   = (__bf16*)alloc((size_t)D_IN_PROJ * D_MODEL * 2);
  __bf16* owb   = (__bf16*)alloc((size_t)D_MODEL * D_INNER * 2);
  float*  zx    = (float*) alloc((size_t)T_TOK * D_IN_PROJ * 4);
  float*  xbc   = (float*) alloc((size_t)T_TOK * CONV_DIM * 4);
  float*  dtb   = (float*) alloc((size_t)T_TOK * N_HEADS * 4);
  float*  dacs  = (float*) alloc((size_t)B_SZ * N_HEADS * NC * CHUNK * 4);
  float*  dalast= (float*) alloc((size_t)B_SZ * N_HEADS * NC * 4);
  float*  states= (float*) alloc((size_t)B_SZ * NC * N_HEADS * D_HEAD * D_STATE * 4);
  float*  prevb = (float*) alloc((size_t)B_SZ * NC * N_HEADS * D_HEAD * D_STATE * 4);
  float*  Ybuf  = (float*) alloc((size_t)T_TOK * D_INNER * 4);
  __bf16* y2b   = (__bf16*)alloc((size_t)T_TOK * D_INNER * 2);
  float*  xmid  = (float*) alloc((size_t)T_TOK * D_MODEL * 4);

  const size_t lds_gemm   = 2 * (64 * 32 + 64 * 32) * 2; // 16KB, double-buffered stages
  const size_t lds_states = (size_t)(D_HEAD * CHUNK + CHUNK * D_STATE) * 2; // 64KB
  const size_t lds_out = (size_t)(CHUNK * D_STATE * 2 + CHUNK * D_HEAD +
                                  D_HEAD * D_STATE + 8 * 16 * 32) * 2 + CHUNK * 4;

  for (int layer = 0; layer < 2; ++layer) {
    const float* xin  = (layer == 0) ? x : xmid;
    float*       xout = (layer == 0) ? xmid : (float*)d_out;
    const float* iw = in_w    + (size_t)layer * D_IN_PROJ * D_MODEL;
    const float* cw = conv_w  + (size_t)layer * CONV_DIM * D_CONV;
    const float* cb = conv_b  + (size_t)layer * CONV_DIM;
    const float* db = dt_bias + (size_t)layer * N_HEADS;
    const float* al = A_log   + (size_t)layer * N_HEADS;
    const float* dv = Dp      + (size_t)layer * N_HEADS;
    const float* gw = gnorm_w + (size_t)layer * D_INNER;
    const float* ow = out_w   + (size_t)layer * D_MODEL * D_INNER;
    const float* nw = norm_w  + (size_t)layer * D_MODEL;

    k_cast_bf16<<<(D_IN_PROJ * D_MODEL + 255) / 256, 256, 0, stream>>>(
        iw, iwb, D_IN_PROJ * D_MODEL);
    k_cast_bf16<<<(D_MODEL * D_INNER + 255) / 256, 256, 0, stream>>>(
        ow, owb, D_MODEL * D_INNER);
    k_rmsnorm_bf16<<<T_TOK, 256, 0, stream>>>(xin, nw, hb, D_MODEL);
    k_gemm_bf16<<<dim3((D_IN_PROJ + 63) / 64, T_TOK / 64), 256, lds_gemm, stream>>>(
        hb, iwb, nullptr, zx, T_TOK, D_IN_PROJ, D_MODEL);
    k_conv_silu<<<dim3((CONV_DIM + 255) / 256, T_TOK), 256, 0, stream>>>(zx, cw, cb, xbc);
    k_dt<<<(T_TOK * N_HEADS + 255) / 256, 256, 0, stream>>>(zx, db, dtb);
    k_cumsum<<<dim3(NC, N_HEADS, B_SZ), CHUNK, 0, stream>>>(dtb, al, dacs, dalast);
    k_chunk_states<<<dim3(N_HEADS, NC, B_SZ), 128, lds_states, stream>>>(
        xbc, dtb, dacs, dalast, states);
    k_state_scan<<<dim3(N_HEADS, B_SZ), 256, 0, stream>>>(states, dalast, prevb);
    k_chunk_output<<<dim3(B_SZ * NC, N_HEADS), 256, lds_out, stream>>>(
        xbc, dtb, dacs, prevb, dv, Ybuf);
    k_gate_rmsnorm_bf16<<<T_TOK, 256, 0, stream>>>(Ybuf, zx, gw, y2b);
    k_gemm_bf16<<<dim3(D_MODEL / 64, T_TOK / 64), 256, lds_gemm, stream>>>(
        y2b, owb, xin, xout, T_TOK, D_MODEL, D_INNER);
  }
}
